// ADS_30313879175331
// MI455X (gfx1250) — matrix operand order repaired and checked by reading, not compiled, not run on hardware
//
#include <hip/hip_runtime.h>
#include <hip/hip_bf16.h>
#include <math.h>

// Problem sizes (fixed by the reference)
#define BB 4
#define TT 16384
#define DD 1024
#define AT 128
#define RR 4
#define M1 (BB * TT)          // 65536 rows into GEMM1
#define M2 (BB * (TT / RR))   // 16384 rows into GEMM2
#define KT1 (DD / 32)         // 32 k-steps of 32

// Trivial ext-vector types (safe inside unions, unlike HIP's uint4/float4)
typedef __bf16       v16bf __attribute__((ext_vector_type(16)));
typedef float        v8f   __attribute__((ext_vector_type(8)));
typedef float        v4f   __attribute__((ext_vector_type(4)));
typedef unsigned int v4u   __attribute__((ext_vector_type(4)));

union Frag {
    v16bf        v;
    unsigned int u[8];
    v4u          q[2];
};

__device__ __forceinline__ unsigned int f2bf1(float f) {
    unsigned int u = __float_as_uint(f);
    u += 0x7fffu + ((u >> 16) & 1u);     // round-to-nearest-even
    return u >> 16;
}
__device__ __forceinline__ unsigned int pk2(float lo, float hi) {
    return f2bf1(lo) | (f2bf1(hi) << 16);
}

__device__ __forceinline__ v8f wmma_bf16(const Frag& a, const Frag& b, v8f c) {
    // (neg_a, A, neg_b, B, c_mod, C, reuse_a, reuse_b)
    return __builtin_amdgcn_wmma_f32_16x16x32_bf16(false, a.v, false, b.v,
                                                   (short)0, c, false, false);
}

// CDNA5 async global->LDS copy (ASYNCcnt-tracked, CDNA5 ISA).
// ldsOff = byte offset within the workgroup LDS allocation (low 32 bits of the
// generic shared-memory address, per the flat-aperture LDS mapping rule).
__device__ __forceinline__ void async_ld_b128(unsigned ldsOff, const void* gptr) {
    asm volatile("global_load_async_to_lds_b128 %0, %1, off"
                 :: "v"(ldsOff), "v"(gptr) : "memory");
}
__device__ __forceinline__ void wait_asynccnt0() {
    asm volatile("s_wait_asynccnt 0x0" ::: "memory");
}

// ---------------------------------------------------------------------------
// Pack a row-major f32 (K x N) weight into bf16 WMMA B-fragment panels:
// fragment index = (nTile*(K/32) + kTile)*32 + lane, 8 dwords each.
// Lane layout (V_WMMA_F32_16X16X32_BF16 B-matrix): lanes 0-15 hold K=k0..k0+7,
// k0+16..k0+23 for column n0+lane; lanes 16-31 hold K=k0+8..k0+15, k0+24..k0+31.
// ---------------------------------------------------------------------------
__global__ void __launch_bounds__(256)
pack_b_kernel(const float* __restrict__ src, unsigned int* __restrict__ dst,
              int K, int N)
{
    int idx = blockIdx.x * blockDim.x + threadIdx.x;   // one per (nt,kt,lane)
    int total = (N / 16) * (K / 32) * 32;
    if (idx >= total) return;
    int lane = idx & 31;
    int kt   = (idx >> 5) % (K / 32);
    int nt   = idx / (32 * (K / 32));
    int n     = nt * 16 + (lane & 15);
    int kbase = kt * 32 + ((lane >> 4) << 3);
    unsigned int* out = dst + (size_t)idx * 8;
#pragma unroll
    for (int j = 0; j < 8; ++j) {
        float lo = src[(size_t)(kbase + 2 * j + ((j >> 2) << 3)) * N + n];
        float hi = src[(size_t)(kbase + 2 * j + ((j >> 2) << 3) + 1) * N + n];
        out[j] = pk2(lo, hi);
    }
}

// ---------------------------------------------------------------------------
// Attention score: s[m] = w2 . tanh(x[m,:]*W1 + b1) + b2   (m = b*T + t)
// Block = 256 threads = 8 waves; each wave owns 16 rows x full ATTN=128.
// All 8 B fragments are loaded into distinct registers first so the compiler
// can emit one load clause + a single wait ahead of the 8 WMMAs.
// ---------------------------------------------------------------------------
__global__ void __launch_bounds__(256)
attn_score_kernel(const float* __restrict__ x, const unsigned int* __restrict__ w1p,
                  const float* __restrict__ b1, const float* __restrict__ w2,
                  const float* __restrict__ b2, float* __restrict__ s)
{
    const int wave = threadIdx.x >> 5;
    const int lane = threadIdx.x & 31;
    const int laneLo = lane & 15, laneHi = lane >> 4;
    const int m0 = blockIdx.x * 128 + wave * 16;
    const float* arow = x + (size_t)(m0 + laneLo) * DD;

    v8f zero = {};
    v8f acc[8];
#pragma unroll
    for (int a = 0; a < 8; ++a) acc[a] = zero;

    const int koff = laneHi * 8;   // A frag: half-wave K offset
    for (int kt = 0; kt < KT1; ++kt) {
        const float* ap = arow + kt * 32 + koff;
        v4f f0 = *(const v4f*)(ap);
        v4f f1 = *(const v4f*)(ap + 4);
        v4f f2 = *(const v4f*)(ap + 16);
        v4f f3 = *(const v4f*)(ap + 20);
        Frag af;
        af.u[0] = pk2(f0.x, f0.y); af.u[1] = pk2(f0.z, f0.w);
        af.u[2] = pk2(f1.x, f1.y); af.u[3] = pk2(f1.z, f1.w);
        af.u[4] = pk2(f2.x, f2.y); af.u[5] = pk2(f2.z, f2.w);
        af.u[6] = pk2(f3.x, f3.y); af.u[7] = pk2(f3.z, f3.w);

        Frag bf[8];
#pragma unroll
        for (int a = 0; a < 8; ++a) {
            const v4u* bp = (const v4u*)(w1p + ((size_t)(a * KT1 + kt) * 32 + lane) * 8);
            bf[a].q[0] = bp[0];
            bf[a].q[1] = bp[1];
        }
#pragma unroll
        for (int a = 0; a < 8; ++a)
            acc[a] = wmma_bf16(af, bf[a], acc[a]);
    }

    // epilogue: tanh, dot with w2, reduce 16 columns per row across half-wave
    float b1v[8], w2v[8];
#pragma unroll
    for (int a = 0; a < 8; ++a) {
        int n = a * 16 + laneLo;
        b1v[a] = b1[n];
        w2v[a] = w2[n];
    }
    const float bias2 = b2[0];
#pragma unroll
    for (int r = 0; r < 8; ++r) {
        float sum = 0.f;
#pragma unroll
        for (int a = 0; a < 8; ++a)
            sum += tanhf(acc[a][r] + b1v[a]) * w2v[a];
#pragma unroll
        for (int off = 1; off < 16; off <<= 1)
            sum += __shfl_xor(sum, off, 32);
        if (laneLo == 0)
            s[m0 + r + laneHi * 8] = sum + bias2;   // lanes 0 / 16 -> rows r / r+8
    }
}

// ---------------------------------------------------------------------------
// Per-batch softmax stats over T: stats[2b] = max, stats[2b+1] = sum(exp(s-max))
// ---------------------------------------------------------------------------
__global__ void __launch_bounds__(256)
softmax_stats_kernel(const float* __restrict__ s, float* __restrict__ stats)
{
    __shared__ float red[256];
    const int b = blockIdx.x;
    const float* sb = s + (size_t)b * TT;
    float mx = -3.0e38f;
    for (int i = threadIdx.x; i < TT; i += 256) mx = fmaxf(mx, sb[i]);
    red[threadIdx.x] = mx;
    __syncthreads();
    for (int o = 128; o > 0; o >>= 1) {
        if (threadIdx.x < o)
            red[threadIdx.x] = fmaxf(red[threadIdx.x], red[threadIdx.x + o]);
        __syncthreads();
    }
    mx = red[0];
    __syncthreads();
    float sm = 0.f;
    for (int i = threadIdx.x; i < TT; i += 256) sm += expf(sb[i] - mx);
    red[threadIdx.x] = sm;
    __syncthreads();
    for (int o = 128; o > 0; o >>= 1) {
        if (threadIdx.x < o)
            red[threadIdx.x] += red[threadIdx.x + o];
        __syncthreads();
    }
    if (threadIdx.x == 0) {
        stats[2 * b]     = mx;
        stats[2 * b + 1] = red[0];
    }
}

// ---------------------------------------------------------------------------
// GEMM1: h[m,n] = gelu( (x[b,perm[t],:] * (1+aw)) . w_embed[:,n] + b_embed[n] )
// stored bf16.  Block: 128 rows (8 waves x 16) x 64 cols (4 WMMA N-tiles).
// B panels double-buffered in LDS via async global->LDS copies: the k+1 panel
// copy overlaps with the current k-step's WMMAs.
// ---------------------------------------------------------------------------
__global__ void __launch_bounds__(256)
embed_gemm_kernel(const float* __restrict__ x, const unsigned int* __restrict__ wp,
                  const float* __restrict__ bias, const int* __restrict__ perm,
                  const float* __restrict__ s, const float* __restrict__ stats,
                  unsigned short* __restrict__ h)
{
    __shared__ v4u ldsB[2][256];   // 2 x (4 tiles x 32 lanes x 32B) = 8 KB
    const int wave = threadIdx.x >> 5;
    const int lane = threadIdx.x & 31;
    const int laneLo = lane & 15, laneHi = lane >> 4;
    const int m0 = blockIdx.x * 128 + wave * 16;
    const int n0 = blockIdx.y * 64;

    const int m  = m0 + laneLo;
    const int b  = m >> 14;            // T = 16384
    const int t  = m & (TT - 1);
    const int pt = perm[t];
    const float mx    = stats[2 * b];
    const float denom = stats[2 * b + 1];
    const float scale = 1.0f + expf(s[(b << 14) + pt] - mx) / denom;
    const float* arow = x + ((size_t)(b << 14) + (size_t)pt) * DD;

    const int stA = threadIdx.x >> 6;   // staging: which N-tile
    const int stR = threadIdx.x & 63;   // which 16B chunk
    const int ntBase = n0 >> 4;

    v8f zero = {};
    v8f acc[4];
#pragma unroll
    for (int a = 0; a < 4; ++a) acc[a] = zero;

    // prologue: fill buffer 0
    {
        const v4u* src = (const v4u*)(wp + ((size_t)((ntBase + stA) * KT1 + 0) * 32) * 8) + stR;
        async_ld_b128((unsigned)(size_t)&ldsB[0][(stA << 6) + stR], src);
        wait_asynccnt0();
        __syncthreads();
    }

    for (int kt = 0; kt < KT1; ++kt) {
        const int buf = kt & 1;
        if (kt + 1 < KT1) {   // start next panel's async copy (overlaps WMMAs)
            const v4u* src = (const v4u*)(wp + ((size_t)((ntBase + stA) * KT1 + kt + 1) * 32) * 8) + stR;
            async_ld_b128((unsigned)(size_t)&ldsB[buf ^ 1][(stA << 6) + stR], src);
        }

        const float* ap = arow + kt * 32 + laneHi * 8;
        v4f f0 = *(const v4f*)(ap);
        v4f f1 = *(const v4f*)(ap + 4);
        v4f f2 = *(const v4f*)(ap + 16);
        v4f f3 = *(const v4f*)(ap + 20);
        Frag af;
        af.u[0] = pk2(f0.x * scale, f0.y * scale); af.u[1] = pk2(f0.z * scale, f0.w * scale);
        af.u[2] = pk2(f1.x * scale, f1.y * scale); af.u[3] = pk2(f1.z * scale, f1.w * scale);
        af.u[4] = pk2(f2.x * scale, f2.y * scale); af.u[5] = pk2(f2.z * scale, f2.w * scale);
        af.u[6] = pk2(f3.x * scale, f3.y * scale); af.u[7] = pk2(f3.z * scale, f3.w * scale);

        Frag bf[4];
#pragma unroll
        for (int a = 0; a < 4; ++a) {
            bf[a].q[0] = ldsB[buf][(a << 6) + (lane << 1)];
            bf[a].q[1] = ldsB[buf][(a << 6) + (lane << 1) + 1];
        }
#pragma unroll
        for (int a = 0; a < 4; ++a)
            acc[a] = wmma_bf16(af, bf[a], acc[a]);

        wait_asynccnt0();
        __syncthreads();
    }

#pragma unroll
    for (int a = 0; a < 4; ++a) {
        const int n = n0 + a * 16 + laneLo;
        const float bn = bias[n];
#pragma unroll
        for (int r = 0; r < 8; ++r) {
            float v = acc[a][r] + bn;
            v = 0.5f * v * (1.0f + erff(v * 0.70710678118f));   // exact gelu
            h[(size_t)(m0 + r + laneHi * 8) * DD + n] = (unsigned short)f2bf1(v);
        }
    }
}

// ---------------------------------------------------------------------------
// Gather: z[b,tt,j] = h[b, tt*4 + (k&3), k>>2],  k = keep_idx[j]
// (implements reshape(B,T/R,R,D).transpose(0,1,3,2)[..., keep_idx])
// ---------------------------------------------------------------------------
__global__ void __launch_bounds__(256)
gather_z_kernel(const unsigned short* __restrict__ h, const int* __restrict__ keep,
                unsigned short* __restrict__ z)
{
    int idx = blockIdx.x * blockDim.x + threadIdx.x;   // M2*DD total
    int j   = idx & (DD - 1);
    int row = idx >> 10;
    int b  = row >> 12;                 // T/R = 4096
    int tt = row & 4095;
    int k  = keep[j];
    z[idx] = h[(size_t)((b << 14) + (tt << 2) + (k & 3)) * DD + (k >> 2)];
}

// ---------------------------------------------------------------------------
// GEMM2: out[row,o] = z[row,:] . w_down[:,o] + b_down[o]   (f32 out)
// Same double-buffered async LDS staging as GEMM1.
// ---------------------------------------------------------------------------
__global__ void __launch_bounds__(256)
down_gemm_kernel(const unsigned short* __restrict__ z, const unsigned int* __restrict__ wp,
                 const float* __restrict__ bias, float* __restrict__ out)
{
    __shared__ v4u ldsB[2][256];
    const int wave = threadIdx.x >> 5;
    const int lane = threadIdx.x & 31;
    const int laneLo = lane & 15, laneHi = lane >> 4;
    const int m0 = blockIdx.x * 128 + wave * 16;
    const int n0 = blockIdx.y * 64;

    const unsigned short* arow = z + (size_t)(m0 + laneLo) * DD;

    const int stA = threadIdx.x >> 6;
    const int stR = threadIdx.x & 63;
    const int ntBase = n0 >> 4;

    v8f zero = {};
    v8f acc[4];
#pragma unroll
    for (int a = 0; a < 4; ++a) acc[a] = zero;

    {
        const v4u* src = (const v4u*)(wp + ((size_t)((ntBase + stA) * KT1 + 0) * 32) * 8) + stR;
        async_ld_b128((unsigned)(size_t)&ldsB[0][(stA << 6) + stR], src);
        wait_asynccnt0();
        __syncthreads();
    }

    for (int kt = 0; kt < KT1; ++kt) {
        const int buf = kt & 1;
        if (kt + 1 < KT1) {
            const v4u* src = (const v4u*)(wp + ((size_t)((ntBase + stA) * KT1 + kt + 1) * 32) * 8) + stR;
            async_ld_b128((unsigned)(size_t)&ldsB[buf ^ 1][(stA << 6) + stR], src);
        }

        const unsigned short* ap = arow + kt * 32 + laneHi * 8;  // bf16, 16B aligned
        Frag af;
        af.q[0] = ((const v4u*)ap)[0];   // K = base .. base+7
        af.q[1] = ((const v4u*)ap)[2];   // K = base+16 .. base+23

        Frag bf[4];
#pragma unroll
        for (int a = 0; a < 4; ++a) {
            bf[a].q[0] = ldsB[buf][(a << 6) + (lane << 1)];
            bf[a].q[1] = ldsB[buf][(a << 6) + (lane << 1) + 1];
        }
#pragma unroll
        for (int a = 0; a < 4; ++a)
            acc[a] = wmma_bf16(af, bf[a], acc[a]);

        wait_asynccnt0();
        __syncthreads();
    }

#pragma unroll
    for (int a = 0; a < 4; ++a) {
        const int n = n0 + a * 16 + laneLo;
        const float bn = bias[n];
#pragma unroll
        for (int r = 0; r < 8; ++r)
            out[(size_t)(m0 + r + laneHi * 8) * DD + n] = acc[a][r] + bn;
    }
}

// ---------------------------------------------------------------------------
extern "C" void kernel_launch(void* const* d_in, const int* in_sizes, int n_in,
                              void* d_out, int out_size, void* d_ws, size_t ws_size,
                              hipStream_t stream)
{
    (void)in_sizes; (void)n_in; (void)out_size; (void)ws_size;
    const float* x       = (const float*)d_in[0];
    const float* w_attn1 = (const float*)d_in[1];
    const float* b_attn1 = (const float*)d_in[2];
    const float* w_attn2 = (const float*)d_in[3];
    const float* b_attn2 = (const float*)d_in[4];
    const float* w_embed = (const float*)d_in[5];
    const float* b_embed = (const float*)d_in[6];
    const float* w_down  = (const float*)d_in[7];
    const float* b_down  = (const float*)d_in[8];
    const int*   perm    = (const int*)d_in[9];
    const int*   keep    = (const int*)d_in[10];
    float* out = (float*)d_out;

    // workspace carve (all regions 256B-aligned by construction)
    char* w = (char*)d_ws;
    float* s = (float*)w;                 w += (size_t)M1 * 4;                       // 256 KB
    float* stats = (float*)w;             w += 256;
    unsigned int* w1p = (unsigned int*)w; w += (size_t)(AT / 16) * KT1 * 32 * 8 * 4; // 256 KB
    unsigned int* wep = (unsigned int*)w; w += (size_t)(DD / 16) * KT1 * 32 * 8 * 4; // 2 MB
    unsigned int* wdp = (unsigned int*)w; w += (size_t)(DD / 16) * KT1 * 32 * 8 * 4; // 2 MB
    unsigned short* h = (unsigned short*)w; w += (size_t)M1 * DD * 2;                // 128 MB
    unsigned short* z = (unsigned short*)w; w += (size_t)M2 * DD * 2;                // 32 MB

    // 1) pack the three weight matrices into WMMA bf16 B-fragment panels
    {
        int tot1 = (AT / 16) * KT1 * 32;
        pack_b_kernel<<<(tot1 + 255) / 256, 256, 0, stream>>>(w_attn1, w1p, DD, AT);
        int tot2 = (DD / 16) * KT1 * 32;
        pack_b_kernel<<<(tot2 + 255) / 256, 256, 0, stream>>>(w_embed, wep, DD, DD);
        pack_b_kernel<<<(tot2 + 255) / 256, 256, 0, stream>>>(w_down, wdp, DD, DD);
    }
    // 2) attention scores
    attn_score_kernel<<<M1 / 128, 256, 0, stream>>>(x, w1p, b_attn1, w_attn2, b_attn2, s);
    // 3) per-batch softmax stats
    softmax_stats_kernel<<<BB, 256, 0, stream>>>(s, stats);
    // 4) fused scale+permute+GEMM+gelu -> bf16 h
    {
        dim3 g(M1 / 128, DD / 64);
        embed_gemm_kernel<<<g, 256, 0, stream>>>(x, wep, b_embed, perm, s, stats, h);
    }
    // 5) keep_idx gather (reshape/transpose fused) -> bf16 z
    gather_z_kernel<<<((size_t)M2 * DD) / 256, 256, 0, stream>>>(h, keep, z);
    // 6) down-projection GEMM -> f32 out
    {
        dim3 g(M2 / 128, DD / 64);
        down_gemm_kernel<<<g, 256, 0, stream>>>(z, wdp, b_down, out);
    }
}
